// RandomConv_14834817040938
// MI455X (gfx1250) — matrix operand order repaired and checked by reading, not compiled, not run on hardware
//
#include <hip/hip_runtime.h>

// CDNA5 / gfx1250 (wave32, WMMA). Per-sample grouped 3x3 conv, replicate pad,
// sigmoid*255. Contraction of length 27 (3 ci x 3 kh x 3 kw) mapped onto one
// v_wmma_f32_16x16x32_f16 per 16-pixel output tile (K padded 27 -> 32).
// - 1/255 input scale folded into the register-resident B matrix
// - sigmoid via native v_rcp_f32
// - tap tables fully compile-time (cndmask on lane-group, no runtime /9, /3)
// - column offset folded into row base: interior tap address = one add
// - column clamps only in the two edge tiles of each row

typedef __attribute__((ext_vector_type(16))) _Float16 v16h;
typedef __attribute__((ext_vector_type(8)))  float    v8f;

#define HW  512
#define CHN 3

__device__ __forceinline__ int iclamp(int v, int lo, int hi) {
  return v < lo ? lo : (v > hi ? hi : v);
}

__device__ __forceinline__ float sig255(float v) {
  // 255 / (1 + e^-v) with native-rate rcp (v_exp_f32 + v_rcp_f32 + mul/add)
  return 255.0f * __builtin_amdgcn_rcpf(1.0f + __expf(-v));
}

template <bool CLAMP_COLS>
__device__ __forceinline__ void conv_tile(const float* __restrict__ xin,
                                          const int (&rowBase)[16],   // no dC
                                          const int (&rowBaseC)[16],  // +dC folded
                                          const int (&dC)[16],
                                          int c0, int p, int g,
                                          v16h b, float* __restrict__ optrRow) {
  const int col = c0 + p;               // A pixel column for row M = p

  // ---- Assemble A: 16 taps per lane. Interior tiles: one add + load + cvt
  // per tap. For a fixed tap, lanes 0..15 read 16 consecutive floats ->
  // coalesced 64B segments with ~9x reuse out of WGP$/L2.
  v16h a;
#pragma unroll
  for (int j = 0; j < 16; ++j) {
    float v;
    if (CLAMP_COLS) {
      int scol = iclamp(col + dC[j], 0, HW - 1);      // replicate pad (edges)
      v = xin[rowBase[j] + scol];
    } else {
      v = xin[rowBaseC[j] + col];
    }
    a[j] = (_Float16)v;
  }

  // ---- One WMMA per 16-pixel tile: D(16x16 f32) = A(16x32) * B(32x16)
  v8f c = {};
  c = __builtin_amdgcn_wmma_f32_16x16x32_f16(
          /*neg_a=*/false, a, /*neg_b=*/false, b,
          /*c_mod=*/(short)0, c, /*reuse_a=*/false, /*reuse_b=*/false);

  // ---- D layout: lane&15 = co, element m -> pixel (c0 + m + 8*g).
  // 8 contiguous floats per useful lane -> two b128 stores.
  if (p < CHN) {
    float* optr = optrRow + c0;         // optrRow already includes +8*g
    float4 lo, hi;
    lo.x = sig255(c[0]);  lo.y = sig255(c[1]);
    lo.z = sig255(c[2]);  lo.w = sig255(c[3]);
    hi.x = sig255(c[4]);  hi.y = sig255(c[5]);
    hi.z = sig255(c[6]);  hi.w = sig255(c[7]);
    *reinterpret_cast<float4*>(optr)     = lo;
    *reinterpret_cast<float4*>(optr + 4) = hi;
  }
}

__global__ __launch_bounds__(256)
void RandomConv_14834817040938_kernel(const float* __restrict__ x,
                                      const float* __restrict__ w,
                                      float* __restrict__ out) {
  const int n    = blockIdx.y;          // sample
  const int rb   = blockIdx.x;          // block of 16 output rows
  const int tid  = threadIdx.x;
  const int wave = tid >> 5;            // 8 waves per workgroup (wave32)
  const int lane = tid & 31;
  const int g    = lane >> 4;           // half-wave group (selects K half)
  const int p    = lane & 15;           // A: pixel row M;  B/D: column N (= co)

  const float* xin = x + (size_t)n * (CHN * HW * HW);
  const float* wn  = w + n * 81;        // weights[n] as [co][k], k = ci*9+kh*3+kw

  // ---- B matrix (32x16 f16, K x N): lane&15 = N(co), element j -> K = 8*g + (j<8?j:j+8)
  // Input scale 1/255 folded in; K>=27 and N>=3 stay zero, which also
  // zero-pads whatever the clamped A-loads fetch for K>=27.
  v16h b = {};
  if (p < CHN) {
#pragma unroll
    for (int j = 0; j < 16; ++j) {
      int K = 8 * g + ((j < 8) ? j : j + 8);
      float wv = (K < 27) ? wn[p * 27 + K] * (1.0f / 255.0f) : 0.0f;
      b[j] = (_Float16)wv;
    }
  }

  // ---- Per-lane tap tables for A (16x32 f16): element j -> K = 8*g + (j<8?j:j+8)
  // Constants for BOTH lane groups are compile-time folded; per-thread select
  // is a single cndmask per entry (no runtime /9, /3 to rematerialize).
  int dR[16], dC[16], chOff[16];
#pragma unroll
  for (int j = 0; j < 16; ++j) {
    const int Ka  = (j < 8) ? j : j + 8;        // K for g==0
    const int Kb  = Ka + 8;                     // K for g==1
    const int Kac = (Ka < 27) ? Ka : 0;         // in-bounds addr; B zeros K>=27
    const int Kbc = (Kb < 27) ? Kb : 0;
    const int ci0 = Kac / 9, ra = Kac - ci0 * 9, kh0 = ra / 3, kw0 = ra - kh0 * 3;
    const int ci1 = Kbc / 9, rb_ = Kbc - ci1 * 9, kh1 = rb_ / 3, kw1 = rb_ - kh1 * 3;
    dR[j]    = g ? (kh1 - 1) : (kh0 - 1);
    dC[j]    = g ? (kw1 - 1) : (kw0 - 1);
    chOff[j] = (g ? ci1 : ci0) * (HW * HW);
  }

  const int r0 = rb * 16 + wave * 2;    // each wave handles 2 output rows
  const int pc = (p < CHN) ? p : 0;     // safe co for pointer precompute

#pragma unroll
  for (int rr = 0; rr < 2; ++rr) {
    const int r = r0 + rr;

    // Row-invariant: clamped source-row bases (replicate pad on rows),
    // plus the column-offset-folded variant for interior tiles.
    int rowBase[16], rowBaseC[16];
#pragma unroll
    for (int j = 0; j < 16; ++j) {
      rowBase[j]  = chOff[j] + iclamp(r + dR[j], 0, HW - 1) * HW;
      rowBaseC[j] = rowBase[j] + dC[j];
    }

    float* optrRow = out + (((size_t)(n * CHN + pc)) * HW + r) * HW + 8 * g;

    // Edge tiles need the column clamp; 30 interior tiles do not.
    conv_tile<true>(xin, rowBase, rowBaseC, dC, 0, p, g, b, optrRow);
    for (int t = 1; t < HW / 16 - 1; ++t)
      conv_tile<false>(xin, rowBase, rowBaseC, dC, t * 16, p, g, b, optrRow);
    conv_tile<true>(xin, rowBase, rowBaseC, dC, HW - 16, p, g, b, optrRow);
  }
}

extern "C" void kernel_launch(void* const* d_in, const int* in_sizes, int n_in,
                              void* d_out, int out_size, void* d_ws, size_t ws_size,
                              hipStream_t stream) {
  const float* x = (const float*)d_in[0];   // [64,3,512,512] f32
  const float* w = (const float*)d_in[1];   // [64,3,3,3,3]  f32
  float* out = (float*)d_out;               // [64,3,512,512] f32

  dim3 grid(HW / 16, 64);   // 32 row-blocks x 64 samples
  dim3 block(256);          // 8 waves (wave32)
  RandomConv_14834817040938_kernel<<<grid, block, 0, stream>>>(x, w, out);
}
